// GraphAttentionLayer_60687887893076
// MI455X (gfx1250) — matrix operand order repaired and checked by reading, not compiled, not run on hardware
//
#include <hip/hip_runtime.h>
#include <cstdint>
#include <cstddef>

typedef __attribute__((ext_vector_type(16))) _Float16 v16h;
typedef __attribute__((ext_vector_type(8)))  float    v8f;

#define N_NODES 8192
#define F_IN    128
#define F_OUT   64
#define LRELU_ALPHA 0.2f
#define NEG_BIG (-9.0e15f)
#define NWAVES  8            // waves per attention block (split over columns)

// Workspace layout (bytes). Total ~4.3 MB.
#define OFF_XF32  ((size_t)0)                       // 8192*64 f32   = 2 MB
#define OFF_F1    ((size_t)2097152)                 // 8192 f32      = 32 KB
#define OFF_F2    ((size_t)(2097152 + 32768))       // 8192 f32      = 32 KB
#define OFF_XF16  ((size_t)(2097152 + 65536))       // 8192*64 f16   = 1 MB
#define OFF_XPACK (OFF_XF16 + (size_t)1048576)      // packed B frags = 1 MB
#define OFF_WPACK (OFF_XPACK + (size_t)1048576)     // packed W frags = 16 KB

// ---------------------------------------------------------------------------
// Repack W (f32, [128][64]) into f16 WMMA B-fragment layout:
// fragment (kb, ct): 32x16 tile. Lane l<16: N=l, K={0..7,16..23}; lane>=16: N=l-16, K={8..15,24..31}.
__global__ void gat_pack_w(const float* __restrict__ W, _Float16* __restrict__ Wpack) {
  int tid = blockIdx.x * blockDim.x + threadIdx.x;
  if (tid >= 512) return;                       // 16 fragments * 32 lanes
  int lane = tid & 31;
  int frag = tid >> 5;                          // kb*4 + ct
  int kb = frag >> 2, ct = frag & 3;
  int n = ct * 16 + (lane & 15);
  int kbase = kb * 32 + 8 * (lane >> 4);
  _Float16* dst = Wpack + ((size_t)frag * 32 + lane) * 16;
#pragma unroll
  for (int k = 0; k < 16; ++k)
    dst[k] = (_Float16)W[(size_t)(kbase + k + (k >= 8 ? 8 : 0)) * F_OUT + n];
}

// ---------------------------------------------------------------------------
// x = input @ W with v_wmma_f32_16x16x32_f16. One wave per 16x16 output tile.
__global__ void __launch_bounds__(32)
gat_gemm_x(const float* __restrict__ in, const _Float16* __restrict__ Wpack,
           float* __restrict__ xf32, _Float16* __restrict__ xf16) {
  const int lane = threadIdx.x & 31;
  const int hi = lane >> 4;
  const int ct = blockIdx.x & 3;
  const int rowbase = (blockIdx.x >> 2) * 16;
  const int myrow = rowbase + (lane & 15);
  // A-fragment (f16 16x32): lanes 0-15 hold K {0..7,16..23}; lanes 16-31 hold {8..15,24..31}
  const float* ap = in + (size_t)myrow * F_IN + 8 * hi;
  const v16h* wp = (const v16h*)Wpack;
  v8f acc = {};
#pragma unroll
  for (int kb = 0; kb < 4; ++kb) {
    float4 a0 = *(const float4*)(ap + kb * 32 + 0);
    float4 a1 = *(const float4*)(ap + kb * 32 + 4);
    float4 a2 = *(const float4*)(ap + kb * 32 + 16);
    float4 a3 = *(const float4*)(ap + kb * 32 + 20);
    v16h af;
    af[0]  = (_Float16)a0.x; af[1]  = (_Float16)a0.y; af[2]  = (_Float16)a0.z; af[3]  = (_Float16)a0.w;
    af[4]  = (_Float16)a1.x; af[5]  = (_Float16)a1.y; af[6]  = (_Float16)a1.z; af[7]  = (_Float16)a1.w;
    af[8]  = (_Float16)a2.x; af[9]  = (_Float16)a2.y; af[10] = (_Float16)a2.z; af[11] = (_Float16)a2.w;
    af[12] = (_Float16)a3.x; af[13] = (_Float16)a3.y; af[14] = (_Float16)a3.z; af[15] = (_Float16)a3.w;
    v16h bf = wp[(size_t)(kb * 4 + ct) * 32 + lane];
    acc = __builtin_amdgcn_wmma_f32_16x16x32_f16(false, af, false, bf, (short)0, acc, false, false);
  }
  // C layout: VGPR r, lanes 0-15 -> M=r ; lanes 16-31 -> M=r+8 ; N = lane%16
#pragma unroll
  for (int r = 0; r < 8; ++r) {
    int row = rowbase + r + 8 * hi;
    int col = ct * 16 + (lane & 15);
    float v = acc[r];
    xf32[(size_t)row * F_OUT + col] = v;
    xf16[(size_t)row * F_OUT + col] = (_Float16)v;
  }
}

// ---------------------------------------------------------------------------
// f1 = x @ a[:64], f2 = x @ a[64:]
__global__ void gat_f1f2(const float* __restrict__ xf32, const float* __restrict__ a,
                         float* __restrict__ f1, float* __restrict__ f2) {
  int r = blockIdx.x * blockDim.x + threadIdx.x;
  if (r >= N_NODES) return;
  const float* xr = xf32 + (size_t)r * F_OUT;
  float s1 = 0.f, s2 = 0.f;
#pragma unroll
  for (int n = 0; n < F_OUT; ++n) {
    float v = xr[n];
    s1 += v * a[n];
    s2 += v * a[F_OUT + n];
  }
  f1[r] = s1;
  f2[r] = s2;
}

// ---------------------------------------------------------------------------
// Repack x(f16) into attention-B fragments: fragment (jb, ct) is the 32x16 tile
// x[jb*32 .. jb*32+31][ct*16 .. ct*16+15], laid out 16 contiguous halves/lane.
__global__ void gat_pack_x(const _Float16* __restrict__ xf16, _Float16* __restrict__ xpack) {
  int tid = blockIdx.x * blockDim.x + threadIdx.x;   // 256*4*32 = 32768
  int lane = tid & 31;
  int ct = (tid >> 5) & 3;
  int jb = tid >> 7;
  if (jb >= 256) return;
  int col = ct * 16 + (lane & 15);
  int kbase = jb * 32 + 8 * (lane >> 4);
  _Float16* dst = xpack + ((size_t)(jb * 4 + ct) * 32 + lane) * 16;
#pragma unroll
  for (int k = 0; k < 16; ++k)
    dst[k] = xf16[(size_t)(kbase + k + (k >= 8 ? 8 : 0)) * F_OUT + col];
}

// ---------------------------------------------------------------------------
// Fused masked-attention + softmax + (attn @ x) + ELU.
// 8 waves per block; wave w owns column-tiles jb = w, w+8, ...
// adj tiles are staged LDS-side with double-buffered async copies
// (global_load_async_to_lds_b128 / s_wait_asynccnt), WMMA accumulation,
// per-wave online softmax, cross-wave merge in LDS.
__global__ void __launch_bounds__(NWAVES * 32)
gat_attn(const int* __restrict__ adj, const _Float16* __restrict__ xpack,
         const float* __restrict__ f1, const float* __restrict__ f2,
         float* __restrict__ out) {
  __shared__ __align__(16) int adjbuf[NWAVES][2][512];   // 32 KB: 2 KB tile x2 per wave
  __shared__ float accs[NWAVES][16][F_OUT];              // 32 KB
  __shared__ float ms[NWAVES][16];
  __shared__ float ls[NWAVES][16];

  const int lane = threadIdx.x & 31;
  const int w    = threadIdx.x >> 5;
  const int hi   = lane >> 4;
  const int rowbase = blockIdx.x * 16;
  const float f1i = f1[rowbase + (lane & 15)];
  const float* f2p = f2 + 8 * hi;
  const v16h*  xp  = (const v16h*)xpack;

  // --- async staging addressing -------------------------------------------
  // chunk c (0..3): lane moves 16B of row (4c + lane>>3), col chunk (lane&7).
  const int* gasync = adj + (size_t)(rowbase + (lane >> 3)) * N_NODES + (lane & 7) * 4;
  const uint32_t lbase = (uint32_t)(uintptr_t)&adjbuf[w][0][0]
                       + (uint32_t)(((lane >> 3) * 32 + (lane & 7) * 4) * 4);
  auto issue_adj = [&](int jbase, int par) {
#pragma unroll
    for (int c = 0; c < 4; ++c) {
      uint64_t ga = (uint64_t)(uintptr_t)(gasync + jbase) +
                    (uint64_t)c * (4ull * N_NODES * sizeof(int));
      uint32_t la = lbase + (uint32_t)par * 2048u + (uint32_t)c * 512u;
      asm volatile("global_load_async_to_lds_b128 %0, %1, off"
                   :: "v"(la), "v"(ga) : "memory");
    }
  };

  float m = NEG_BIG, lsum = 0.f;
  v8f acc[4] = {{}, {}, {}, {}};

  issue_adj(w * 32, 0);                       // prologue: stage first tile
  int par = 0;

  for (int jb = w; jb < 256; jb += NWAVES) {
    const int jbase = jb * 32;
    // stage next tile into the other buffer, then wait for current tile
    if (jb + NWAVES < 256) {
      issue_adj((jb + NWAVES) * 32, par ^ 1);
      asm volatile("s_wait_asynccnt 0x4" ::: "memory");
    } else {
      asm volatile("s_wait_asynccnt 0x0" ::: "memory");
    }

    const int* bufp = &adjbuf[w][par][0];     // row r at bufp + r*32 (ints)
    int4 A0 = *(const int4*)(bufp + (lane & 15) * 32 + 8 * hi);
    int4 A1 = *(const int4*)(bufp + (lane & 15) * 32 + 8 * hi + 4);
    int4 A2 = *(const int4*)(bufp + (lane & 15) * 32 + 8 * hi + 16);
    int4 A3 = *(const int4*)(bufp + (lane & 15) * 32 + 8 * hi + 20);
    float4 G0 = *(const float4*)(f2p + jbase);
    float4 G1 = *(const float4*)(f2p + jbase + 4);
    float4 G2 = *(const float4*)(f2p + jbase + 16);
    float4 G3 = *(const float4*)(f2p + jbase + 20);

    int   av[16] = {A0.x, A0.y, A0.z, A0.w, A1.x, A1.y, A1.z, A1.w,
                    A2.x, A2.y, A2.z, A2.w, A3.x, A3.y, A3.z, A3.w};
    float gv[16] = {G0.x, G0.y, G0.z, G0.w, G1.x, G1.y, G1.z, G1.w,
                    G2.x, G2.y, G2.z, G2.w, G3.x, G3.y, G3.z, G3.w};

    float s[16];
    float tmax = -3.0e38f;
#pragma unroll
    for (int k = 0; k < 16; ++k) {
      float sv = f1i + gv[k];
      sv = sv > 0.f ? sv : LRELU_ALPHA * sv;       // leaky relu
      sv = av[k] > 0 ? sv : NEG_BIG;               // adjacency mask
      s[k] = sv;
      tmax = fmaxf(tmax, sv);
    }
    // combine the two half-wave K-subsets of each row (wave32 shuffle)
    tmax = fmaxf(tmax, __shfl_xor(tmax, 16, 32));
    float mnew  = fmaxf(m, tmax);
    float alpha = __expf(m - mnew);
    float tsum = 0.f;
    v16h pa;
#pragma unroll
    for (int k = 0; k < 16; ++k) {
      float p = __expf(s[k] - mnew);
      tsum += p;
      pa[k] = (_Float16)p;
    }
    tsum += __shfl_xor(tsum, 16, 32);
    lsum = lsum * alpha + tsum;
    m = mnew;

    // per-output-row rescale: C-layout row M = r + 8*hi, state lives in lane M
    float alr[8];
#pragma unroll
    for (int r = 0; r < 8; ++r) alr[r] = __shfl(alpha, r + 8 * hi, 32);

#pragma unroll
    for (int t = 0; t < 4; ++t) {
      v16h b = xp[(size_t)(jb * 4 + t) * 32 + lane];
#pragma unroll
      for (int r = 0; r < 8; ++r) acc[t][r] *= alr[r];
      acc[t] = __builtin_amdgcn_wmma_f32_16x16x32_f16(false, pa, false, b, (short)0,
                                                      acc[t], false, false);
    }
    par ^= 1;
  }

  // ---- dump per-wave partial state to LDS --------------------------------
  if (lane < 16) {
    ms[w][lane] = m;
    ls[w][lane] = lsum;
  }
#pragma unroll
  for (int t = 0; t < 4; ++t)
#pragma unroll
    for (int r = 0; r < 8; ++r)
      accs[w][r + 8 * hi][t * 16 + (lane & 15)] = acc[t][r];
  __syncthreads();

  // ---- merge NWAVES partial softmax states, finalize, store --------------
  for (int e = threadIdx.x; e < 16 * F_OUT; e += NWAVES * 32) {
    int i = e >> 6;          // local row
    int c = e & 63;          // column
    float M = NEG_BIG;
#pragma unroll
    for (int q = 0; q < NWAVES; ++q) M = fmaxf(M, ms[q][i]);
    float L = 0.f, S = 0.f;
#pragma unroll
    for (int q = 0; q < NWAVES; ++q) {
      float sc = __expf(ms[q][i] - M);
      L += ls[q][i] * sc;
      S += accs[q][i][c] * sc;
    }
    float v = S / L;
    v = v > 0.f ? v : (__expf(v) - 1.0f);          // ELU
    out[(size_t)(rowbase + i) * F_OUT + c] = v;
  }
}

// ---------------------------------------------------------------------------
extern "C" void kernel_launch(void* const* d_in, const int* in_sizes, int n_in,
                              void* d_out, int out_size, void* d_ws, size_t ws_size,
                              hipStream_t stream) {
  (void)in_sizes; (void)n_in; (void)out_size; (void)ws_size;
  const float* input = (const float*)d_in[0];
  const int*   adj   = (const int*)d_in[1];
  const float* W     = (const float*)d_in[2];
  const float* a     = (const float*)d_in[3];
  float* out = (float*)d_out;

  char* ws = (char*)d_ws;
  float*    xf32  = (float*)(ws + OFF_XF32);
  float*    f1    = (float*)(ws + OFF_F1);
  float*    f2    = (float*)(ws + OFF_F2);
  _Float16* xf16  = (_Float16*)(ws + OFF_XF16);
  _Float16* xpack = (_Float16*)(ws + OFF_XPACK);
  _Float16* wpack = (_Float16*)(ws + OFF_WPACK);

  hipLaunchKernelGGL(gat_pack_w, dim3(2),    dim3(256), 0, stream, W, wpack);
  hipLaunchKernelGGL(gat_gemm_x, dim3(2048), dim3(32),  0, stream, input, wpack, xf32, xf16);
  hipLaunchKernelGGL(gat_f1f2,   dim3(32),   dim3(256), 0, stream, xf32, a, f1, f2);
  hipLaunchKernelGGL(gat_pack_x, dim3(128),  dim3(256), 0, stream, xf16, xpack);
  hipLaunchKernelGGL(gat_attn,   dim3(512),  dim3(NWAVES * 32), 0, stream,
                     adj, xpack, f1, f2, out);
}
